// ConditionalOutputLayer_59545426591814
// MI455X (gfx1250) — matrix operand order repaired and checked by reading, not compiled, not run on hardware
//
#include <hip/hip_runtime.h>

#define BATCH 16384
#define FDIM  2048
#define DDIM  256
#define NEXP  8
#define NKT   (FDIM / 32)      // 64 k-tiles of 32
#define NNT   (DDIM / 16)      // 16 n-tiles of 16

typedef __bf16 v16bf __attribute__((ext_vector_type(16)));
typedef __bf16 v8bf  __attribute__((ext_vector_type(8)));
typedef float  v8f   __attribute__((ext_vector_type(8)));

// float -> bf16 round-to-nearest-even (bit trick)
__device__ __forceinline__ __bf16 f2bf(float x) {
  union { float f; unsigned u; } v; v.f = x;
  unsigned r = v.u + 0x7FFFu + ((v.u >> 16) & 1u);
  union { unsigned short s; __bf16 b; } o; o.s = (unsigned short)(r >> 16);
  return o.b;
}

// ---------------- bucketing kernels ----------------
__global__ void k_zero(int* p, int n) {
  int i = blockIdx.x * blockDim.x + threadIdx.x;
  if (i < n) p[i] = 0;
}

__global__ void k_count(const int* __restrict__ ids, int* __restrict__ counts) {
  int i = blockIdx.x * blockDim.x + threadIdx.x;
  if (i < BATCH) atomicAdd(&counts[ids[i]], 1);
}

__global__ void k_scan(const int* __restrict__ counts, int* __restrict__ offsets) {
  if (threadIdx.x == 0) {
    int s = 0;
    for (int e = 0; e < NEXP; ++e) { offsets[e] = s; s += counts[e]; }
  }
}

__global__ void k_place(const int* __restrict__ ids, const int* __restrict__ offsets,
                        int* __restrict__ cursors, int* __restrict__ lists) {
  int i = blockIdx.x * blockDim.x + threadIdx.x;
  if (i < BATCH) {
    int e = ids[i];
    int p = atomicAdd(&cursors[e], 1);
    lists[offsets[e] + p] = i;
  }
}

// ---------------- W pre-swizzle: f32 [e][k][d] -> fragment-ready bf16 ----------------
// Tile (e,kt,nt) -> 512 bf16: lane L holds B[k][n] for n = nt*16+(L&15),
// k = kt*32 + ((L<16)?0:16) + j, j=0..15  (contiguous k halves, same order as the A reads in k_gemm)
__global__ void k_swizzleW(const float* __restrict__ W, __bf16* __restrict__ Wf) {
  int tid  = blockIdx.x * blockDim.x + threadIdx.x;   // NEXP*NKT*NNT*32 threads
  int lane = tid & 31;
  int tile = tid >> 5;                                // 0 .. 8191
  int nt   = tile & (NNT - 1);
  int kt   = (tile >> 4) & (NKT - 1);
  int e    = tile >> 10;
  int n    = nt * 16 + (lane & 15);
  int kb   = kt * 32 + ((lane < 16) ? 0 : 16);
  const float* src = W + ((size_t)e * FDIM + kb) * DDIM + n;
  v16bf v;
  #pragma unroll
  for (int j = 0; j < 16; ++j) v[j] = f2bf(src[(size_t)j * DDIM]);
  *(v16bf*)(Wf + (size_t)tile * 512 + lane * 16) = v;
}

// ---------------- grouped GEMM: 64 rows x 256 cols per block ----------------
__launch_bounds__(256)
__global__ void k_gemm(const float* __restrict__ X, const __bf16* __restrict__ Wf,
                       const float* __restrict__ bias,
                       const int* __restrict__ counts, const int* __restrict__ offsets,
                       const int* __restrict__ lists, float* __restrict__ Y)
{
  __shared__ __bf16 lA[64 * 64];        // A tile [m][k], 64 rows, K=64 per barrier
  __shared__ int    rIdx[64];

  const int e        = blockIdx.x >> 8;        // 256 tiles per expert
  const int tile     = blockIdx.x & 255;
  const int cnt      = counts[e];
  const int rowsBase = tile * 64;
  if (rowsBase >= cnt) return;                 // uniform per block
  const int off = offsets[e];

  const int t    = threadIdx.x;
  const int lane = t & 31;
  const int wv   = t >> 5;                     // wave id 0..7

  if (t < 64) {
    int g = rowsBase + t;
    rIdx[t] = (g < cnt) ? lists[off + g] : -1;
  }
  __syncthreads();

  // 4 row tiles x 2 col tiles of accumulators per wave
  v8f acc[4][2] = {};

  const int mA    = lane & 15;
  const int klo_a = (lane < 16) ? 0 : 16;      // A: contiguous K halves
  const int n0    = wv * 32;

  // A staging: 4 threads per row, 16 consecutive k each
  const int srow = t >> 2;
  const int skc  = (t & 3) * 16;
  const int srO  = rIdx[srow];
  const float* xsrc = X + (size_t)(srO < 0 ? 0 : srO) * FDIM + skc;

  const size_t tbase = (size_t)e * NKT * NNT;  // tile index base for this expert

  for (int it = 0; it < FDIM / 64; ++it) {
    const int k0 = it * 64;
    // ---- stage A (16 floats -> 16 bf16 per thread) ----
    {
      float4 u0 = make_float4(0.f, 0.f, 0.f, 0.f), u1 = u0, u2 = u0, u3 = u0;
      if (srO >= 0) {
        u0 = *(const float4*)(xsrc + k0);
        u1 = *(const float4*)(xsrc + k0 + 4);
        u2 = *(const float4*)(xsrc + k0 + 8);
        u3 = *(const float4*)(xsrc + k0 + 12);
        if (k0 + 64 < FDIM) __builtin_prefetch(xsrc + k0 + 64, 0, 3);
      }
      v8bf p0, p1;
      p0[0] = f2bf(u0.x); p0[1] = f2bf(u0.y); p0[2] = f2bf(u0.z); p0[3] = f2bf(u0.w);
      p0[4] = f2bf(u1.x); p0[5] = f2bf(u1.y); p0[6] = f2bf(u1.z); p0[7] = f2bf(u1.w);
      p1[0] = f2bf(u2.x); p1[1] = f2bf(u2.y); p1[2] = f2bf(u2.z); p1[3] = f2bf(u2.w);
      p1[4] = f2bf(u3.x); p1[5] = f2bf(u3.y); p1[6] = f2bf(u3.z); p1[7] = f2bf(u3.w);
      *(v8bf*)&lA[srow * 64 + skc]     = p0;
      *(v8bf*)&lA[srow * 64 + skc + 8] = p1;
    }
    __syncthreads();

    #pragma unroll
    for (int s = 0; s < 2; ++s) {
      const int kt = it * 2 + s;
      // B fragments: coalesced, fragment-ready from global (L2-resident)
      const __bf16* bp = Wf + (tbase + (size_t)kt * NNT + wv * 2) * 512 + lane * 16;
      const v16bf b0 = *(const v16bf*)bp;
      const v16bf b1 = *(const v16bf*)(bp + 512);

      #pragma unroll
      for (int mt = 0; mt < 4; ++mt) {
        const int mrow = mt * 16 + mA;
        const v8bf a_lo = *(const v8bf*)&lA[mrow * 64 + s * 32 + klo_a];
        const v8bf a_hi = *(const v8bf*)&lA[mrow * 64 + s * 32 + klo_a + 8];
        const v16bf a = __builtin_shufflevector(a_lo, a_hi,
                          0,1,2,3,4,5,6,7,8,9,10,11,12,13,14,15);
        acc[mt][0] = __builtin_amdgcn_wmma_f32_16x16x32_bf16(false, a, false, b0,
                        (short)0, acc[mt][0], false, false);
        acc[mt][1] = __builtin_amdgcn_wmma_f32_16x16x32_bf16(false, a, false, b1,
                        (short)0, acc[mt][1], false, false);
      }
    }
    __syncthreads();
  }

  // ---- epilogue: bias add + scatter to gathered rows ----
  const float* bp = bias + (size_t)e * DDIM;
  const int mbase = (lane < 16) ? 0 : 8;       // C/D layout: VGPR j -> M=j or j+8
  const int ncol  = lane & 15;
  const int nc0 = n0 + ncol, nc1 = n0 + 16 + ncol;
  const float bb0 = bp[nc0], bb1 = bp[nc1];

  #pragma unroll
  for (int mt = 0; mt < 4; ++mt) {
    #pragma unroll
    for (int j = 0; j < 8; ++j) {
      int r = rIdx[mt * 16 + mbase + j];
      if (r >= 0) {
        Y[(size_t)r * DDIM + nc0] = acc[mt][0][j] + bb0;
        Y[(size_t)r * DDIM + nc1] = acc[mt][1][j] + bb1;
      }
    }
  }
}

// ---------------- launch ----------------
extern "C" void kernel_launch(void* const* d_in, const int* in_sizes, int n_in,
                              void* d_out, int out_size, void* d_ws, size_t ws_size,
                              hipStream_t stream) {
  const float* X    = (const float*)d_in[0];   // [B,F]
  const float* W    = (const float*)d_in[1];   // [E+1,F,D]
  const float* bias = (const float*)d_in[2];   // [E+1,D]
  const int*   ids  = (const int*)d_in[3];     // [B,1]
  float*       Y    = (float*)d_out;           // [B,D]

  // ws layout: [0, 8MB) swizzled W (bf16), then bucketing ints
  __bf16* Wf   = (__bf16*)d_ws;
  int* wsi     = (int*)((char*)d_ws + (size_t)NEXP * NKT * NNT * 512 * sizeof(__bf16));
  int* counts  = wsi;          // 16 ints
  int* offsets = wsi + 16;     // 16 ints
  int* cursors = wsi + 32;     // 16 ints
  int* lists   = wsi + 48;     // BATCH ints

  k_zero    <<<1, 64, 0, stream>>>(wsi, 48);
  k_count   <<<BATCH / 256, 256, 0, stream>>>(ids, counts);
  k_scan    <<<1, 32, 0, stream>>>(counts, offsets);
  k_place   <<<BATCH / 256, 256, 0, stream>>>(ids, offsets, cursors, lists);
  k_swizzleW<<<(NEXP * NKT * NNT * 32) / 256, 256, 0, stream>>>(W, Wf);
  k_gemm    <<<NEXP * (BATCH / 64), 256, 0, stream>>>(X, Wf, bias, counts, offsets, lists, Y);
}
